// Model_68332929679536
// MI455X (gfx1250) — matrix operand order repaired and checked by reading, not compiled, not run on hardware
//
#include <hip/hip_runtime.h>
#include <math.h>
#include <stdint.h>

// ---------------- types ----------------
typedef __bf16 bf16_t;
typedef __attribute__((ext_vector_type(16))) __bf16 v16bf;
typedef __attribute__((ext_vector_type(8)))  float  v8f;
typedef __attribute__((ext_vector_type(4)))  unsigned int u32x4;

// ---------------- model constants ----------------
#define BB 8
#define CC 3
#define HH 384
#define WW 384
#define PP 16
#define DD 768
#define HEADS 8
#define HDIM 96
#define DEPTH 12
#define NPATCH 576
#define NTOK 577
#define NPAD 608            // 577 padded to multiple of 32
#define DFF 3072
#define MTOK (BB*NTOK)      // 4616
#define SCALE_ATTN 0.03608439182435161f   // 768^-0.5

__device__ __forceinline__ void wait_asynccnt0() {
#if __has_builtin(__builtin_amdgcn_s_wait_asynccnt)
    __builtin_amdgcn_s_wait_asynccnt(0);
#else
    asm volatile("s_wait_asynccnt 0x0" ::: "memory");
#endif
}

// ================= elementwise kernels =================

__global__ void k_cvt_bf16(const float* __restrict__ s, bf16_t* __restrict__ d, long long n) {
    long long i = (long long)blockIdx.x * blockDim.x + threadIdx.x;
    if (i < n) d[i] = (bf16_t)s[i];
}

__global__ void k_convw_T(const float* __restrict__ w, bf16_t* __restrict__ d) {
    int i = blockIdx.x * blockDim.x + threadIdx.x;   // over 768*768
    if (i >= DD * DD) return;
    int k = i / DD, n = i % DD;
    d[(long long)k * DD + n] = (bf16_t)w[(long long)n * DD + k];
}

__global__ void k_patchify(const float* __restrict__ x, bf16_t* __restrict__ out) {
    long long i = (long long)blockIdx.x * blockDim.x + threadIdx.x;
    if (i >= (long long)BB * NPATCH * DD) return;
    int col = (int)(i % DD);
    long long m = i / DD;
    int p = (int)(m % NPATCH);
    int b = (int)(m / NPATCH);
    int ph = p / 24, pw = p % 24;
    int c = col / (PP * PP);
    int r = col % (PP * PP);
    int py = r / PP, px = r % PP;
    long long src = (((long long)b * CC + c) * HH + (ph * PP + py)) * WW + (pw * PP + px);
    out[i] = (bf16_t)x[src];
}

__global__ void k_assemble_h(const float* __restrict__ ptmp, const float* __restrict__ cls,
                             const float* __restrict__ pos, float* __restrict__ h) {
    long long i = (long long)blockIdx.x * blockDim.x + threadIdx.x;
    if (i >= (long long)BB * NTOK * DD) return;
    int dcol = (int)(i % DD);
    long long t = i / DD;
    int n = (int)(t % NTOK);
    int b = (int)(t / NTOK);
    float v = (n == 0) ? cls[dcol] : ptmp[((long long)b * NPATCH + (n - 1)) * DD + dcol];
    h[i] = v + pos[(long long)n * DD + dcol];
}

__global__ void k_layernorm_bf16(const float* __restrict__ x, const float* __restrict__ g,
                                 const float* __restrict__ be, bf16_t* __restrict__ y) {
    int row = blockIdx.x;
    const float* xr = x + (long long)row * DD;
    __shared__ float red[256];
    int tid = threadIdx.x;
    float s = 0.f;
    for (int j = tid; j < DD; j += 256) s += xr[j];
    red[tid] = s; __syncthreads();
    for (int o = 128; o > 0; o >>= 1) { if (tid < o) red[tid] += red[tid + o]; __syncthreads(); }
    float mean = red[0] / (float)DD; __syncthreads();
    float v = 0.f;
    for (int j = tid; j < DD; j += 256) { float d0 = xr[j] - mean; v += d0 * d0; }
    red[tid] = v; __syncthreads();
    for (int o = 128; o > 0; o >>= 1) { if (tid < o) red[tid] += red[tid + o]; __syncthreads(); }
    float rstd = rsqrtf(red[0] / (float)DD + 1e-5f);
    bf16_t* yr = y + (long long)row * DD;
    for (int j = tid; j < DD; j += 256)
        yr[j] = (bf16_t)((xr[j] - mean) * rstd * g[j] + be[j]);
}

// [b*577+n][h*96+d] fp32 -> [(b*8+h)][n][d] bf16, zero pad n in [577,608)
__global__ void k_permute_qv(const float* __restrict__ src, bf16_t* __restrict__ dst) {
    long long i = (long long)blockIdx.x * blockDim.x + threadIdx.x;
    if (i >= (long long)BB * HEADS * NPAD * HDIM) return;
    int d0 = (int)(i % HDIM);
    long long t = i / HDIM;
    int n = (int)(t % NPAD);
    int bh = (int)(t / NPAD);
    int b = bh / HEADS, hh = bh % HEADS;
    float v = 0.f;
    if (n < NTOK) v = src[((long long)(b * NTOK + n)) * DD + hh * HDIM + d0];
    dst[i] = (bf16_t)v;
}

// softmax over rows of energy [64*577][577], write attn bf16 [64*577][608] zero-padded
__global__ void k_softmax(const float* __restrict__ e, bf16_t* __restrict__ attn) {
    long long row = blockIdx.x;
    const float* er = e + row * NTOK;
    bf16_t* ar = attn + row * NPAD;
    __shared__ float red[256];
    int tid = threadIdx.x;
    float m = -3.4e38f;
    for (int j = tid; j < NTOK; j += 256) m = fmaxf(m, er[j]);
    red[tid] = m; __syncthreads();
    for (int o = 128; o > 0; o >>= 1) { if (tid < o) red[tid] = fmaxf(red[tid], red[tid + o]); __syncthreads(); }
    float gm = red[0] * SCALE_ATTN; __syncthreads();
    float s = 0.f;
    for (int j = tid; j < NTOK; j += 256) s += __expf(er[j] * SCALE_ATTN - gm);
    red[tid] = s; __syncthreads();
    for (int o = 128; o > 0; o >>= 1) { if (tid < o) red[tid] += red[tid + o]; __syncthreads(); }
    float inv = 1.f / red[0];
    for (int j = tid; j < NPAD; j += 256)
        ar[j] = (j < NTOK) ? (bf16_t)(__expf(er[j] * SCALE_ATTN - gm) * inv) : (bf16_t)0.f;
}

// [(b*8+h)][n][d] fp32 -> [b*577+n][h*96+d] bf16
__global__ void k_permute_o(const float* __restrict__ o, bf16_t* __restrict__ omat) {
    long long i = (long long)blockIdx.x * blockDim.x + threadIdx.x;
    if (i >= (long long)MTOK * DD) return;
    int col = (int)(i % DD);
    long long t = i / DD;
    int n = (int)(t % NTOK);
    int b = (int)(t / NTOK);
    int hh = col / HDIM, d0 = col % HDIM;
    omat[i] = (bf16_t)o[(((long long)(b * HEADS + hh)) * NTOK + n) * HDIM + d0];
}

// mean-pool over tokens + final LN + classifier head (tiny)
__global__ void k_pool_head(const float* __restrict__ h, const float* __restrict__ g,
                            const float* __restrict__ be, const float* __restrict__ hw,
                            const float* __restrict__ hb, float* __restrict__ out) {
    int b = blockIdx.x, tid = threadIdx.x;
    __shared__ float pooled[DD];
    __shared__ float red[256];
    for (int d0 = tid; d0 < DD; d0 += 256) {
        float s = 0.f;
        const float* hp = h + ((long long)b * NTOK) * DD + d0;
        for (int n = 0; n < NTOK; n++) s += hp[(long long)n * DD];
        pooled[d0] = s / (float)NTOK;
    }
    __syncthreads();
    float s = 0.f;
    for (int j = tid; j < DD; j += 256) s += pooled[j];
    red[tid] = s; __syncthreads();
    for (int o = 128; o > 0; o >>= 1) { if (tid < o) red[tid] += red[tid + o]; __syncthreads(); }
    float mean = red[0] / (float)DD; __syncthreads();
    float v = 0.f;
    for (int j = tid; j < DD; j += 256) { float d0 = pooled[j] - mean; v += d0 * d0; }
    red[tid] = v; __syncthreads();
    for (int o = 128; o > 0; o >>= 1) { if (tid < o) red[tid] += red[tid + o]; __syncthreads(); }
    float rstd = rsqrtf(red[0] / (float)DD + 1e-5f);
    __syncthreads();
    for (int j = tid; j < DD; j += 256) pooled[j] = (pooled[j] - mean) * rstd * g[j] + be[j];
    __syncthreads();
    if (tid < 10) {
        float s2 = hb[tid];
        for (int d0 = 0; d0 < DD; d0++) s2 += pooled[d0] * hw[d0 * 10 + tid];
        out[b * 10 + tid] = s2;
    }
}

// ================= WMMA bf16 GEMM =================
// C[M,N] = A[M,K](bf16) x B[K,N](bf16) (+bias)(+resid)(+gelu), batched via blockIdx.z.
// 256 threads = 8 wave32 (4x2), block tile 128x128, K staged 32-wide through LDS.
// Interior blocks: double-buffered CDNA5 async global->LDS DMA (ASYNCcnt)
// overlapping the WMMA pipe. Edge/transB blocks: guarded VGPR staging.
__global__ __launch_bounds__(256)
void k_gemm_bf16(const bf16_t* __restrict__ A, const bf16_t* __restrict__ Bm,
                 const float* __restrict__ bias, const float* __restrict__ resid,
                 float* __restrict__ outF, bf16_t* __restrict__ outB,
                 int M, int N, int K, int lda, int ldb, int ldc,
                 long long strA, long long strB, long long strC,
                 int transB, int gelu)
{
    __shared__ __align__(16) bf16_t sA[2][128 * 32];   // [buf][row][k] row stride 32
    __shared__ __align__(16) bf16_t sB[2][32 * 128];   // [buf][k][n]   k stride 128
    const int tid  = threadIdx.x;
    const int lane = tid & 31;
    const int wave = tid >> 5;
    const int wm = wave & 3;        // 4 waves along M
    const int wn = wave >> 2;       // 2 waves along N
    const int m0 = blockIdx.y * 128;
    const int n0 = blockIdx.x * 128;
    const long long zb = blockIdx.z;
    const bf16_t* Ab = A + zb * strA;
    const bf16_t* Bb = Bm + zb * strB;

    v8f acc[2][4];
    const v8f vzero = {0.f, 0.f, 0.f, 0.f, 0.f, 0.f, 0.f, 0.f};
    #pragma unroll
    for (int i = 0; i < 2; i++)
        #pragma unroll
        for (int j = 0; j < 4; j++) acc[i][j] = vzero;

    const int arow = tid >> 1;          // 0..127
    const int aks  = (tid & 1) * 16;    // 0 or 16
    const int bkr  = tid >> 3;          // 0..31
    const int bnc  = (tid & 7) * 16;    // 0..112
    const int tnn  = tid >> 1;          // transB: n within tile
    const int tks  = (tid & 1) * 16;

    // uniform per block: fully-covered tile, no transpose => async fast path
    const bool interior = ((m0 + 128) <= M) && ((n0 + 128) <= N) && !transB;
    const unsigned ldsAad[2] = { (unsigned)(uintptr_t)&sA[0][arow * 32 + aks],
                                 (unsigned)(uintptr_t)&sA[1][arow * 32 + aks] };
    const unsigned ldsBad[2] = { (unsigned)(uintptr_t)&sB[0][bkr * 128 + bnc],
                                 (unsigned)(uintptr_t)&sB[1][bkr * 128 + bnc] };

    // Issue async DMA global->LDS for one 32-wide K chunk (no VGPR round-trip).
    // INST_OFFSET applies to both global and LDS addresses.
    auto issue = [&](int kk, int buf) {
        unsigned long long ga =
            (unsigned long long)(uintptr_t)(Ab + (long long)(m0 + arow) * lda + kk + aks);
        asm volatile("global_load_async_to_lds_b128 %0, %1, off"
                     :: "v"(ldsAad[buf]), "v"(ga) : "memory");
        asm volatile("global_load_async_to_lds_b128 %0, %1, off offset:16"
                     :: "v"(ldsAad[buf]), "v"(ga) : "memory");
        unsigned long long gb =
            (unsigned long long)(uintptr_t)(Bb + (long long)(kk + bkr) * ldb + n0 + bnc);
        asm volatile("global_load_async_to_lds_b128 %0, %1, off"
                     :: "v"(ldsBad[buf]), "v"(gb) : "memory");
        asm volatile("global_load_async_to_lds_b128 %0, %1, off offset:16"
                     :: "v"(ldsBad[buf]), "v"(gb) : "memory");
    };

    // 8 WMMAs on one staged K chunk.
    // A frag (16-bit 16x32): lanes 0-15 rows M=0..15 (K 0-7 then 16-23),
    // lanes 16-31 same rows (K 8-15 then 24-31) -> two contiguous b128 reads.
    // B frag (32x16): lane%16 = N, elements index K in the same order as A -> strided LDS reads.
    auto compute = [&](const bf16_t* cA, const bf16_t* cB) {
        const int r  = lane & 15;
        const int hi = lane >> 4;
        v16bf afr[2];
        #pragma unroll
        for (int mt = 0; mt < 2; mt++) {
            int rb = wm * 32 + mt * 16 + r;
            union { u32x4 u[2]; v16bf v; } ua;
            ua.u[0] = *(const u32x4*)(cA + rb * 32 + hi * 8);
            ua.u[1] = *(const u32x4*)(cA + rb * 32 + 16 + hi * 8);
            afr[mt] = ua.v;
        }
        #pragma unroll
        for (int nt = 0; nt < 4; nt++) {
            union { u32x4 u[2]; v16bf v; } ub;
            const bf16_t* bp = cB + wn * 64 + nt * 16 + (lane & 15);
            #pragma unroll
            for (int s = 0; s < 16; s++) {
                int kq = (s < 8) ? ((lane >> 4) * 8 + s) : (16 + (lane >> 4) * 8 + (s - 8));
                ub.v[s] = bp[kq * 128];
            }
            #pragma unroll
            for (int mt = 0; mt < 2; mt++)
                acc[mt][nt] = __builtin_amdgcn_wmma_f32_16x16x32_bf16(
                    false, afr[mt], false, ub.v, (short)0, acc[mt][nt], false, false);
        }
    };

    if (interior) {
        // double-buffered: DMA of chunk k+1 overlaps WMMA on chunk k
        issue(0, 0);
        wait_asynccnt0();
        __syncthreads();
        int cur = 0;
        for (int k0 = 0; k0 < K; k0 += 32) {
            if (k0 + 32 < K) issue(k0 + 32, cur ^ 1);
            compute(&sA[cur][0], &sB[cur][0]);
            wait_asynccnt0();
            __syncthreads();
            cur ^= 1;
        }
    } else {
        for (int k0 = 0; k0 < K; k0 += 32) {
            // ---- guarded VGPR staging (edge tiles / transB) ----
            {
                int gm = m0 + arow;
                u32x4 u0 = {0u, 0u, 0u, 0u}, u1 = {0u, 0u, 0u, 0u};
                if (gm < M) {
                    const u32x4* p = (const u32x4*)(Ab + (long long)gm * lda + k0 + aks);
                    u0 = p[0]; u1 = p[1];
                    if (k0 + 32 < K)
                        __builtin_prefetch(Ab + (long long)gm * lda + k0 + 32 + aks, 0, 1);
                }
                u32x4* d = (u32x4*)(&sA[0][arow * 32 + aks]);
                d[0] = u0; d[1] = u1;
            }
            if (!transB) {
                int gn = n0 + bnc;
                u32x4 u0 = {0u, 0u, 0u, 0u}, u1 = {0u, 0u, 0u, 0u};
                if (gn + 16 <= N) {
                    const u32x4* p = (const u32x4*)(Bb + (long long)(k0 + bkr) * ldb + gn);
                    u0 = p[0]; u1 = p[1];
                } else if (gn < N) {
                    union { u32x4 u[2]; bf16_t h[16]; } t;
                    t.u[0] = u0; t.u[1] = u1;
                    for (int j = 0; j < 16; j++)
                        if (gn + j < N) t.h[j] = Bb[(long long)(k0 + bkr) * ldb + gn + j];
                    u0 = t.u[0]; u1 = t.u[1];
                }
                u32x4* d = (u32x4*)(&sB[0][bkr * 128 + bnc]);
                d[0] = u0; d[1] = u1;
            } else {
                int gn = n0 + tnn;
                union { u32x4 u[2]; bf16_t h[16]; } t;
                t.u[0] = (u32x4){0u, 0u, 0u, 0u};
                t.u[1] = (u32x4){0u, 0u, 0u, 0u};
                if (gn < N) {
                    const u32x4* p = (const u32x4*)(Bb + (long long)gn * ldb + k0 + tks);
                    t.u[0] = p[0]; t.u[1] = p[1];
                }
                #pragma unroll
                for (int j = 0; j < 16; j++) sB[0][(tks + j) * 128 + tnn] = t.h[j];
            }
            __syncthreads();
            compute(&sA[0][0], &sB[0][0]);
            __syncthreads();
        }
    }

    // ---- epilogue: C vgpr i -> row = hi*8 + i, col = lane&15 within 16x16 tile ----
    const int r  = lane & 15;
    const int hi = lane >> 4;
    #pragma unroll
    for (int mt = 0; mt < 2; mt++) {
        #pragma unroll
        for (int nt = 0; nt < 4; nt++) {
            #pragma unroll
            for (int i = 0; i < 8; i++) {
                int rr = m0 + wm * 32 + mt * 16 + hi * 8 + i;
                int cc = n0 + wn * 64 + nt * 16 + r;
                if (rr < M && cc < N) {
                    float v = acc[mt][nt][i];
                    if (bias)  v += bias[cc];
                    long long off = zb * strC + (long long)rr * ldc + cc;
                    if (resid) v += resid[off];
                    if (gelu)  v = 0.5f * v * (1.f + erff(v * 0.70710678118f));
                    if (outB) outB[off] = (bf16_t)v;
                    else      outF[off] = v;
                }
            }
        }
    }
}

// ================= host orchestration =================
extern "C" void kernel_launch(void* const* d_in, const int* in_sizes, int n_in,
                              void* d_out, int out_size, void* d_ws, size_t ws_size,
                              hipStream_t stream)
{
    (void)in_sizes; (void)n_in; (void)out_size; (void)ws_size;
    const float* x      = (const float*)d_in[0];
    const float* conv_w = (const float*)d_in[1];
    const float* conv_b = (const float*)d_in[2];
    const float* cls    = (const float*)d_in[3];
    const float* pos    = (const float*)d_in[4];
    const float* ln1_g  = (const float*)d_in[5];
    const float* ln1_b  = (const float*)d_in[6];
    const float* wq     = (const float*)d_in[7];
    const float* bq     = (const float*)d_in[8];
    const float* wv     = (const float*)d_in[9];
    const float* bv     = (const float*)d_in[10];
    const float* wo     = (const float*)d_in[11];
    const float* bo     = (const float*)d_in[12];
    const float* ln2_g  = (const float*)d_in[13];
    const float* ln2_b  = (const float*)d_in[14];
    const float* w1     = (const float*)d_in[15];
    const float* b1     = (const float*)d_in[16];
    const float* w2     = (const float*)d_in[17];
    const float* b2     = (const float*)d_in[18];
    const float* lnf_g  = (const float*)d_in[19];
    const float* lnf_b  = (const float*)d_in[20];
    const float* head_w = (const float*)d_in[21];
    const float* head_b = (const float*)d_in[22];
    float* out = (float*)d_out;

    char* p = (char*)d_ws;
    auto carve = [&](size_t bytes) -> char* {
        char* r = p; p += (bytes + 255) & ~(size_t)255; return r;
    };
    bf16_t* convT   = (bf16_t*)carve((size_t)DD * DD * 2);
    bf16_t* wq_bf   = (bf16_t*)carve((size_t)DEPTH * DD * DD * 2);
    bf16_t* wv_bf   = (bf16_t*)carve((size_t)DEPTH * DD * DD * 2);
    bf16_t* wo_bf   = (bf16_t*)carve((size_t)DEPTH * DD * DD * 2);
    bf16_t* w1_bf   = (bf16_t*)carve((size_t)DEPTH * DD * DFF * 2);
    bf16_t* w2_bf   = (bf16_t*)carve((size_t)DEPTH * DFF * DD * 2);
    bf16_t* patches = (bf16_t*)carve((size_t)BB * NPATCH * DD * 2);
    float*  ptmp    = (float*)carve((size_t)BB * NPATCH * DD * 4);
    float*  hbuf    = (float*)carve((size_t)MTOK * DD * 4);
    bf16_t* ybf     = (bf16_t*)carve((size_t)MTOK * DD * 2);
    float*  qtmp    = (float*)carve((size_t)MTOK * DD * 4);
    float*  vtmp    = (float*)carve((size_t)MTOK * DD * 4);
    bf16_t* qp      = (bf16_t*)carve((size_t)BB * HEADS * NPAD * HDIM * 2);
    bf16_t* vp      = (bf16_t*)carve((size_t)BB * HEADS * NPAD * HDIM * 2);
    float*  energy  = (float*)carve((size_t)BB * HEADS * NTOK * NTOK * 4);
    bf16_t* attn    = (bf16_t*)carve((size_t)BB * HEADS * NTOK * NPAD * 2);
    float*  obuf    = (float*)carve((size_t)BB * HEADS * NTOK * HDIM * 4);
    bf16_t* omat    = (bf16_t*)carve((size_t)MTOK * DD * 2);
    bf16_t* fbf     = (bf16_t*)carve((size_t)MTOK * DFF * 2);

    auto cvt = [&](const float* s, bf16_t* d, long long n) {
        k_cvt_bf16<<<dim3((unsigned)((n + 255) / 256)), 256, 0, stream>>>(s, d, n);
    };
    auto gemm = [&](const bf16_t* A, const bf16_t* Bm, const float* bias, const float* resid,
                    float* oF, bf16_t* oB, int M, int N, int K, int lda, int ldb, int ldc,
                    long long sA, long long sB, long long sC, int nb, int tB, int ge) {
        dim3 grid((N + 127) / 128, (M + 127) / 128, nb);
        k_gemm_bf16<<<grid, 256, 0, stream>>>(A, Bm, bias, resid, oF, oB,
                                              M, N, K, lda, ldb, ldc, sA, sB, sC, tB, ge);
    };

    // weights -> bf16 (156 MB, fits in 192 MB L2)
    cvt(wq, wq_bf, (long long)DEPTH * DD * DD);
    cvt(wv, wv_bf, (long long)DEPTH * DD * DD);
    cvt(wo, wo_bf, (long long)DEPTH * DD * DD);
    cvt(w1, w1_bf, (long long)DEPTH * DD * DFF);
    cvt(w2, w2_bf, (long long)DEPTH * DFF * DD);
    k_convw_T<<<(DD * DD + 255) / 256, 256, 0, stream>>>(conv_w, convT);

    // patch embed
    long long npx = (long long)BB * NPATCH * DD;
    k_patchify<<<(unsigned)((npx + 255) / 256), 256, 0, stream>>>(x, patches);
    gemm(patches, convT, conv_b, nullptr, ptmp, nullptr,
         BB * NPATCH, DD, DD, DD, DD, DD, 0, 0, 0, 1, 0, 0);
    long long nh = (long long)BB * NTOK * DD;
    k_assemble_h<<<(unsigned)((nh + 255) / 256), 256, 0, stream>>>(ptmp, cls, pos, hbuf);

    for (int L = 0; L < DEPTH; L++) {
        k_layernorm_bf16<<<MTOK, 256, 0, stream>>>(hbuf, ln1_g + L * DD, ln1_b + L * DD, ybf);
        gemm(ybf, wq_bf + (size_t)L * DD * DD, bq + L * DD, nullptr, qtmp, nullptr,
             MTOK, DD, DD, DD, DD, DD, 0, 0, 0, 1, 0, 0);
        gemm(ybf, wv_bf + (size_t)L * DD * DD, bv + L * DD, nullptr, vtmp, nullptr,
             MTOK, DD, DD, DD, DD, DD, 0, 0, 0, 1, 0, 0);
        long long nqv = (long long)BB * HEADS * NPAD * HDIM;
        k_permute_qv<<<(unsigned)((nqv + 255) / 256), 256, 0, stream>>>(qtmp, qp);
        k_permute_qv<<<(unsigned)((nqv + 255) / 256), 256, 0, stream>>>(vtmp, vp);
        // energy = q @ q^T  (k = q, faithful to source), batched over 64 (b,h)
        gemm(qp, qp, nullptr, nullptr, energy, nullptr,
             NTOK, NTOK, HDIM, HDIM, HDIM, NTOK,
             (long long)NPAD * HDIM, (long long)NPAD * HDIM, (long long)NTOK * NTOK,
             BB * HEADS, 1, 0);
        k_softmax<<<BB * HEADS * NTOK, 256, 0, stream>>>(energy, attn);
        // o = attn @ v (K padded to 608 with zeros)
        gemm(attn, vp, nullptr, nullptr, obuf, nullptr,
             NTOK, HDIM, NPAD, NPAD, HDIM, HDIM,
             (long long)NTOK * NPAD, (long long)NPAD * HDIM, (long long)NTOK * HDIM,
             BB * HEADS, 0, 0);
        long long no = (long long)MTOK * DD;
        k_permute_o<<<(unsigned)((no + 255) / 256), 256, 0, stream>>>(obuf, omat);
        gemm(omat, wo_bf + (size_t)L * DD * DD, bo + L * DD, hbuf, hbuf, nullptr,
             MTOK, DD, DD, DD, DD, DD, 0, 0, 0, 1, 0, 0);
        k_layernorm_bf16<<<MTOK, 256, 0, stream>>>(hbuf, ln2_g + L * DD, ln2_b + L * DD, ybf);
        gemm(ybf, w1_bf + (size_t)L * DD * DFF, b1 + L * DFF, nullptr, nullptr, fbf,
             MTOK, DFF, DD, DD, DFF, DFF, 0, 0, 0, 1, 0, 1);
        gemm(fbf, w2_bf + (size_t)L * DFF * DD, b2 + L * DD, hbuf, hbuf, nullptr,
             MTOK, DD, DFF, DFF, DD, DD, 0, 0, 0, 1, 0, 0);
    }

    k_pool_head<<<BB, 256, 0, stream>>>(hbuf, lnf_g, lnf_b, head_w, head_b, out);
}
